// SparsePatternGenerator_11596411699561
// MI455X (gfx1250) — compile-verified, matrix operand order checked by reading, not run on hardware
//
#include <hip/hip_runtime.h>
#include <hip/hip_bf16.h>
#include <math.h>

typedef __attribute__((ext_vector_type(16))) _Float16 v16h;
typedef __attribute__((ext_vector_type(8)))  float    v8f;

#define WINDOW_RADIUS 256
#define GLOBAL_TOKENS 64
#define RANDOM_BLOCKS 8
#define BIG_IDX 0x7fffffff
#define NEG_INF (-__builtin_inff())

// ---------------------------------------------------------------------------
// K1: per-row top-8 of rand_scores. One wave32 per row. Streaming float4
// loads (the 256MB read that sets the roofline) + global_prefetch_b8.
// Register-resident sorted top-8 per lane, then an 8-round shfl_xor
// tournament merge. Tie-break: equal value -> lower column index (lax.top_k).
// ---------------------------------------------------------------------------
__global__ void topk8_kernel(const float* __restrict__ scores,
                             int* __restrict__ randIdx, int S)
{
    const int wave = threadIdx.x >> 5;
    const int lane = threadIdx.x & 31;
    const int row  = blockIdx.x * 8 + wave;
    if (row >= S) return;
    const float* rp = scores + (size_t)row * S;

    float t[8]; int ix[8];
    #pragma unroll
    for (int j = 0; j < 8; ++j) { t[j] = NEG_INF; ix[j] = BIG_IDX; }

    const int iters = S / 128;                 // 32 lanes * float4
    for (int it = 0; it < iters; ++it) {
        const int c = it * 128 + lane * 4;
        if (it + 2 < iters)
            __builtin_prefetch(rp + c + 256, 0, 1);   // global_prefetch_b8
        const float4 v = *(const float4*)(rp + c);
        float vv[4] = {v.x, v.y, v.z, v.w};
        #pragma unroll
        for (int q = 0; q < 4; ++q) {
            const float val = vv[q];
            const int   idx = c + q;
            // fast reject against current 8th-best
            if (!((val > t[7]) || (val == t[7] && idx < ix[7]))) continue;
            #pragma unroll
            for (int j = 7; j >= 0; --j) {
                const bool b = (val > t[j]) || (val == t[j] && idx < ix[j]);
                if (b) {
                    if (j < 7) { t[j + 1] = t[j]; ix[j + 1] = ix[j]; }
                    t[j] = val; ix[j] = idx;
                }
            }
        }
    }

    // 8-round cross-lane tournament: pop global max each round.
    int myOut = 0;
    #pragma unroll 1
    for (int r = 0; r < 8; ++r) {
        float bv = t[0]; int bi = ix[0]; int bl = lane;
        #pragma unroll
        for (int off = 16; off >= 1; off >>= 1) {
            const float ov = __shfl_xor(bv, off, 32);
            const int   oi = __shfl_xor(bi, off, 32);
            const int   ol = __shfl_xor(bl, off, 32);
            if (ov > bv || (ov == bv && oi < bi)) { bv = ov; bi = oi; bl = ol; }
        }
        if (lane == bl) {                      // winner lane pops its head
            #pragma unroll
            for (int j = 0; j < 7; ++j) { t[j] = t[j + 1]; ix[j] = ix[j + 1]; }
            t[7] = NEG_INF; ix[7] = BIG_IDX;
        }
        if (lane == r) myOut = bi;
    }
    if (lane < 8) randIdx[(size_t)row * 8 + lane] = myOut;
}

// ---------------------------------------------------------------------------
// K2: mask materialization (float 0/1, float4 coalesced stores) fused with
// row-count reduction on the matrix pipe: counts = mask_tile(16x32 f16) x
// ones(32x16) accumulated in f32 via v_wmma_f32_16x16x32_f16. Lane (l&15)
// owns row M for the whole tile, so its 8 random indices live in registers
// and the f16 A operand is built directly in the ISA VGPR layout:
//   lane<16 : elements 0..7 -> K=0..7,  8..15 -> K=16..23
//   lane>=16: elements 0..7 -> K=8..15, 8..15 -> K=24..31
// One wave per block => EXEC all-ones at every WMMA (ISA requirement).
// ---------------------------------------------------------------------------
__global__ void mask_count_kernel(const int* __restrict__ randIdx,
                                  float* __restrict__ maskOut,
                                  int* __restrict__ counts,
                                  int S, int chunkCols)
{
    const int lane     = threadIdx.x;          // blockDim.x == 32
    const int rowBase  = blockIdx.x * 16;
    const int colStart = blockIdx.y * chunkCols;
    const int M        = lane & 15;
    const int row      = rowBase + M;
    const int e0       = (lane >= 16) ? 8 : 0;

    int r[8];
    #pragma unroll
    for (int q = 0; q < 8; ++q) r[q] = randIdx[(size_t)row * 8 + q];
    const int grow = (row < GLOBAL_TOKENS) ? 1 : 0;

    v16h ones;
    #pragma unroll
    for (int j = 0; j < 16; ++j) ones[j] = (_Float16)1.0f;

    v8f acc = {};

    const int ktiles = chunkCols / 32;
    for (int kt = 0; kt < ktiles; ++kt) {
        const int colBase = colStart + kt * 32;
        v16h a;
        float f[16];
        #pragma unroll
        for (int e = 0; e < 16; ++e) {
            const int col = colBase + ((e < 8) ? (e0 + e) : (e0 + e + 8));
            int mm = grow | ((col < GLOBAL_TOKENS) ? 1 : 0);
            const int d = row - col;
            mm |= ((unsigned)(d + WINDOW_RADIUS) <= 2u * WINDOW_RADIUS) ? 1 : 0;
            #pragma unroll
            for (int q = 0; q < 8; ++q) mm |= (col == r[q]) ? 1 : 0;
            a[e] = mm ? (_Float16)1.0f : (_Float16)0.0f;
            f[e] = mm ? 1.0f : 0.0f;
        }
        // mask stores: two 32B runs per lane, 16B aligned
        const size_t b0 = (size_t)row * S + (size_t)colBase + e0;
        *(float4*)(maskOut + b0)      = make_float4(f[0],  f[1],  f[2],  f[3]);
        *(float4*)(maskOut + b0 + 4)  = make_float4(f[4],  f[5],  f[6],  f[7]);
        *(float4*)(maskOut + b0 + 16) = make_float4(f[8],  f[9],  f[10], f[11]);
        *(float4*)(maskOut + b0 + 20) = make_float4(f[12], f[13], f[14], f[15]);

        acc = __builtin_amdgcn_wmma_f32_16x16x32_f16(
                  false, a, false, ones, (short)0, acc, false, false);
    }

    // C/D layout: vgpr r on lanes 0..15 = (M=r, N=lane); lanes 16..31 = M=r+8.
    if (lane == 0) {
        #pragma unroll
        for (int m2 = 0; m2 < 8; ++m2)
            atomicAdd(&counts[rowBase + m2], (int)(acc[m2] + 0.5f));
    } else if (lane == 16) {
        #pragma unroll
        for (int m2 = 0; m2 < 8; ++m2)
            atomicAdd(&counts[rowBase + 8 + m2], (int)(acc[m2] + 0.5f));
    }
}

// ---------------------------------------------------------------------------
// K3: single-block exclusive scan of row counts -> row_offsets (int to ws,
// float to d_out). offsI[S] = total_connections.
// ---------------------------------------------------------------------------
__global__ void scan_kernel(const int* __restrict__ counts,
                            int* __restrict__ offsI,
                            float* __restrict__ offsF, int S)
{
    __shared__ int sums[1024];
    const int t = threadIdx.x;
    const int rowsPer = (S + 1023) / 1024;
    const int begin = t * rowsPer;
    int s = 0;
    for (int i = 0; i < rowsPer; ++i) {
        const int rr = begin + i;
        if (rr < S) s += counts[rr];
    }
    sums[t] = s;
    __syncthreads();
    for (int off = 1; off < 1024; off <<= 1) {
        const int add = (t >= off) ? sums[t - off] : 0;
        __syncthreads();
        sums[t] += add;
        __syncthreads();
    }
    int run = (t == 0) ? 0 : sums[t - 1];
    for (int i = 0; i < rowsPer; ++i) {
        const int rr = begin + i;
        if (rr < S) {
            offsI[rr] = run;
            offsF[rr] = (float)run;
            run += counts[rr];
        }
    }
    if (t == 1023) { offsI[S] = sums[1023]; offsF[S] = (float)sums[1023]; }
}

// ---------------------------------------------------------------------------
// K4: CSR compaction in ascending-column order (matches jnp.nonzero). One
// wave per row; ballot + popc rank within each 32-column strip.
// ---------------------------------------------------------------------------
__global__ void compact_kernel(const int* __restrict__ randIdx,
                               const int* __restrict__ offsI,
                               float* __restrict__ colOut, int S)
{
    const int wave = threadIdx.x >> 5;
    const int lane = threadIdx.x & 31;
    const int row  = blockIdx.x * 8 + wave;
    if (row >= S) return;
    int r[8];
    #pragma unroll
    for (int q = 0; q < 8; ++q) r[q] = randIdx[(size_t)row * 8 + q];
    const int grow = (row < GLOBAL_TOKENS) ? 1 : 0;
    const int base = offsI[row];
    int run = 0;
    for (int cb = 0; cb < S; cb += 32) {
        const int col = cb + lane;
        int mm = grow | ((col < GLOBAL_TOKENS) ? 1 : 0);
        const int d = row - col;
        mm |= ((unsigned)(d + WINDOW_RADIUS) <= 2u * WINDOW_RADIUS) ? 1 : 0;
        #pragma unroll
        for (int q = 0; q < 8; ++q) mm |= (col == r[q]) ? 1 : 0;
        const unsigned b = (unsigned)__ballot(mm != 0);
        if (mm) {
            const int rank = __popc(b & ((1u << lane) - 1u));
            colOut[(size_t)base + run + rank] = (float)col;
        }
        run += __popc(b);
    }
}

// ---------------------------------------------------------------------------
// K5: row_counts (float) + sparsity, placed after the data-dependent
// col_indices region (total read on-device from offsI[S]).
// ---------------------------------------------------------------------------
__global__ void tail_kernel(const int* __restrict__ counts,
                            const int* __restrict__ offsI,
                            float* __restrict__ out, int S)
{
    const int rr = blockIdx.x * 256 + threadIdx.x;
    const int total = offsI[S];
    const size_t base = (size_t)S * S + (size_t)S + 1 + (size_t)total;
    if (rr < S) out[base + rr] = (float)counts[rr];
    if (rr == 0) out[base + S] = (float)total / ((float)S * (float)S);
}

extern "C" void kernel_launch(void* const* d_in, const int* in_sizes, int n_in,
                              void* d_out, int out_size, void* d_ws, size_t ws_size,
                              hipStream_t stream)
{
    const float* scores = (const float*)d_in[1];            // rand_scores [S,S]
    const int S = (int)(sqrt((double)in_sizes[1]) + 0.5);   // 8192

    // workspace layout (ints)
    int* randIdx = (int*)d_ws;                  // S*8
    int* counts  = randIdx + (size_t)S * 8;     // S
    int* offsI   = counts + S;                  // S+1

    float* out     = (float*)d_out;
    float* maskOut = out;                       // S*S
    float* offsF   = out + (size_t)S * S;       // S+1
    float* colOut  = offsF + (S + 1);           // total_connections

    hipMemsetAsync(counts, 0, (size_t)S * sizeof(int), stream);

    topk8_kernel<<<dim3((S + 7) / 8), dim3(256), 0, stream>>>(scores, randIdx, S);

    const int CHUNKS = 16;                      // column chunks per row tile
    const int chunkCols = S / CHUNKS;           // 512, multiple of 32
    mask_count_kernel<<<dim3(S / 16, CHUNKS), dim3(32), 0, stream>>>(
        randIdx, maskOut, counts, S, chunkCols);

    scan_kernel<<<dim3(1), dim3(1024), 0, stream>>>(counts, offsI, offsF, S);

    compact_kernel<<<dim3((S + 7) / 8), dim3(256), 0, stream>>>(randIdx, offsI, colOut, S);

    tail_kernel<<<dim3((S + 255) / 256), dim3(256), 0, stream>>>(counts, offsI, out, S);
}
